// BilinearAttention_57337813402066
// MI455X (gfx1250) — matrix operand order repaired and checked by reading, not compiled, not run on hardware
//
#include <hip/hip_runtime.h>

// Flash-attention forward, B=16, Tq=Tk=2048, D=1024, fp32 in/out.
//   Pass 1 (memory-bound): transpose+convert k -> kT:[B,Tk,D]bf16,
//     v -> vT:[B,D,Tk]bf16 into d_ws (128 MB scratch).
//   Pass 2 (compute-bound): transposed-tile flash attention.
//     Per block: 16 q rows, 8 waves, 256-wide K tiles.
//     S^T = K * Q^T  (k on M axis, q on N axis)  -> softmax over k is
//     an IN-LANE reduction (+1 cross-half shuffle), mask is one b64 load,
//     O^T = V^T * P^T accumulated in fp32, normalized, stored as b128s.
//     All WMMA operands are aligned 16B loads, zero in-loop converts.

#define TQ   2048
#define TK   2048
#define DDIM 1024
#define NB   16
#define NEG_INF_F (-1.0e9f)

typedef __bf16 bf16;
typedef __attribute__((ext_vector_type(16))) bf16  v16bf;
typedef __attribute__((ext_vector_type(8)))  float v8f;

union BV {
    v16bf v;
    uint4 u[2];
    bf16  e[16];
};

// ---------------- Pass 1: tiled transpose + f32->bf16 ----------------------
// in : [gridDim.z][R][C] f32, out : [gridDim.z][C][R] bf16, 64x64 tiles.
__global__ __launch_bounds__(256) void transpose_cvt_kernel(
    const float* __restrict__ in, bf16* __restrict__ out, int R, int C)
{
    __shared__ bf16 lt[64][72];   // [col][row], row stride 144B (16B mult)

    const int b  = blockIdx.z;
    const int c0 = blockIdx.x * 64;
    const int r0 = blockIdx.y * 64;
    const float* ip = in  + ((size_t)b * R + r0) * C + c0;
    bf16*        op = out + ((size_t)b * C + c0) * R + r0;

    const int tc = threadIdx.x & 63;   // column within tile (coalesced reads)
    const int tr = threadIdx.x >> 6;   // 0..3
#pragma unroll
    for (int p = 0; p < 16; ++p) {
        int r = p * 4 + tr;
        lt[tc][r] = (bf16)ip[(size_t)r * C + tc];
    }
    __syncthreads();

    const int oc   = threadIdx.x >> 2;        // output row (= input col) 0..63
    const int rseg = (threadIdx.x & 3) * 16;  // 16 contiguous r per thread
    uint4 w0 = *(const uint4*)(&lt[oc][rseg]);
    uint4 w1 = *(const uint4*)(&lt[oc][rseg + 8]);
    *(uint4*)(op + (size_t)oc * R + rseg)     = w0;   // 2x global_store_b128
    *(uint4*)(op + (size_t)oc * R + rseg + 8) = w1;
}

// ---------------- Pass 2: transposed-tile flash attention ------------------
__global__ __launch_bounds__(256) void fa_fwd_kernel(
    const float* __restrict__ q,
    const bf16*  __restrict__ kT,    // [B][Tk][D]
    const bf16*  __restrict__ vT,    // [B][D][Tk]
    const unsigned char* __restrict__ mask,
    float* __restrict__ out)
{
    const int tid  = threadIdx.x;
    const int wv   = tid >> 5;        // wave 0..7
    const int lane = tid & 31;
    const int l16  = lane & 15;       // this lane's q column
    const int hi   = lane >> 4;

    const int b  = blockIdx.y;
    const int q0 = blockIdx.x * 16;

    const float* qb = q + ((size_t)b * TQ + q0) * DDIM;
    const bf16*  kb = kT + (size_t)b * TK * DDIM;
    const bf16*  vb = vT + (size_t)b * DDIM * TK;
    const unsigned char* mb = mask + (size_t)b * TQ * TK + (size_t)q0 * TK;
    float* ob = out + ((size_t)b * TQ + q0) * DDIM;

    __shared__ bf16  qtile[16][1032];   // row stride 2064B (16B mult)
    __shared__ float redmax[8][16];     // [wave][q]
    __shared__ float redsum[8][16];
    __shared__ bf16  pbuf[16][264];     // P[q][k] 16 x 256 (+8), stride 528B

    // Q f32 -> bf16 into LDS once (coalesced float4 reads, uint2 LDS stores)
    for (int idx = tid; idx < 16 * 256; idx += 256) {
        int row  = idx >> 8;
        int col4 = (idx & 255) * 4;
        float4 f = *(const float4*)(qb + (size_t)row * DDIM + col4);
        union { uint2 u; bf16 e[4]; } t;
        t.e[0] = (bf16)f.x; t.e[1] = (bf16)f.y;
        t.e[2] = (bf16)f.z; t.e[3] = (bf16)f.w;
        *(uint2*)(&qtile[row][col4]) = t.u;
    }
    __syncthreads();

    // per-lane softmax state for q column l16 (2 scalars total)
    float m_run = -3.0e38f, l_run = 0.0f;
    v8f oacc[8];   // O^T: subtile s -> d rows wv*128+s*16+(r+8*hi), col q=l16
#pragma unroll
    for (int s = 0; s < 8; ++s) oacc[s] = 0.0f;

    const int krow = wv * 16 + l16;   // wave's A-row (k) within first 128

    for (int kt = 0; kt < TK; kt += 256) {
        if (kt + 256 < TK) {   // global_prefetch_b8 of next tile's K rows
            __builtin_prefetch(kb + (size_t)(kt + 256 + krow) * DDIM, 0, 0);
            __builtin_prefetch(kb + (size_t)(kt + 384 + krow) * DDIM, 0, 0);
        }

        // ----- S^T = K * Q^T : two 16(k)x16(q) sub-tiles per wave --------
        const bf16* arow0 = kb + (size_t)(kt + krow) * DDIM;        // g=0
        const bf16* arow1 = arow0 + (size_t)128 * DDIM;             // g=1
        v8f sacc0 = 0.0f, sacc1 = 0.0f;
#pragma unroll 4
        for (int d0 = 0; d0 < DDIM; d0 += 32) {
            BV bq;  // B = Q^T: halves d0+hi*8..+7 and d0+16+hi*8..+7
            bq.u[0] = *(const uint4*)(&qtile[l16][d0 + hi * 8]);
            bq.u[1] = *(const uint4*)(&qtile[l16][d0 + hi * 8 + 16]);
            BV a0, a1;  // A = K rows: halves d0+hi*8..+7 and d0+16+hi*8..+7
            const bf16* p0 = arow0 + d0 + hi * 8;
            const bf16* p1 = arow1 + d0 + hi * 8;
            a0.u[0] = *(const uint4*)(p0);
            a0.u[1] = *(const uint4*)(p0 + 16);
            a1.u[0] = *(const uint4*)(p1);
            a1.u[1] = *(const uint4*)(p1 + 16);
            sacc0 = __builtin_amdgcn_wmma_f32_16x16x32_bf16(
                false, a0.v, false, bq.v, (short)0, sacc0, false, false);
            sacc1 = __builtin_amdgcn_wmma_f32_16x16x32_bf16(
                false, a1.v, false, bq.v, (short)0, sacc1, false, false);
        }

        // ----- mask: 8 consecutive k bytes per lane per group ------------
        union { uint2 u; unsigned char c[8]; } mk0, mk1;
        const unsigned char* mrow =
            mb + (size_t)l16 * TK + kt + wv * 16 + 8 * hi;
        mk0.u = *(const uint2*)(mrow);
        mk1.u = *(const uint2*)(mrow + 128);
#pragma unroll
        for (int r = 0; r < 8; ++r) {
            if (mk0.c[r]) sacc0[r] = NEG_INF_F;
            if (mk1.c[r]) sacc1[r] = NEG_INF_F;
        }

        // ----- softmax over k: in-lane + 1 cross-half shuffle + LDS ------
        float mx = fmaxf(sacc0[0], sacc1[0]);
#pragma unroll
        for (int r = 1; r < 8; ++r)
            mx = fmaxf(mx, fmaxf(sacc0[r], sacc1[r]));
        mx = fmaxf(mx, __shfl_xor(mx, 16, 32));
        if (hi == 0) redmax[wv][l16] = mx;
        __syncthreads();

        float tmax = redmax[0][l16];
#pragma unroll
        for (int w2 = 1; w2 < 8; ++w2) tmax = fmaxf(tmax, redmax[w2][l16]);
        float mn    = fmaxf(m_run, tmax);
        float alpha = __expf(m_run - mn);
        m_run = mn;

        union { uint4 u; bf16 e[8]; } pk0, pk1;
        float psum = 0.0f;
#pragma unroll
        for (int r = 0; r < 8; ++r) {
            float p = __expf(sacc0[r] - mn);
            psum += p; pk0.e[r] = (bf16)p;
        }
#pragma unroll
        for (int r = 0; r < 8; ++r) {
            float p = __expf(sacc1[r] - mn);
            psum += p; pk1.e[r] = (bf16)p;
        }
        // P[q][k]: 8 consecutive k per lane per group -> one b128 store
        *(uint4*)(&pbuf[l16][wv * 16 + 8 * hi])       = pk0.u;
        *(uint4*)(&pbuf[l16][128 + wv * 16 + 8 * hi]) = pk1.u;

        psum += __shfl_xor(psum, 16, 32);
        if (hi == 0) redsum[wv][l16] = psum;
        __syncthreads();   // pbuf + redsum visible

        float ts = 0.0f;
#pragma unroll
        for (int w2 = 0; w2 < 8; ++w2) ts += redsum[w2][l16];
        l_run = l_run * alpha + ts;
#pragma unroll
        for (int s = 0; s < 8; ++s)
#pragma unroll
            for (int r = 0; r < 8; ++r) oacc[s][r] *= alpha;

        // ----- O^T += V^T * P^T (wave owns d rows wv*128..+127) ----------
#pragma unroll
        for (int kc = 0; kc < 8; ++kc) {
            BV bp;  // B = P^T: halves kc*32+hi*8..+7 and kc*32+16+hi*8..+7
            bp.u[0] = *(const uint4*)(&pbuf[l16][kc * 32 + hi * 8]);
            bp.u[1] = *(const uint4*)(&pbuf[l16][kc * 32 + hi * 8 + 16]);
#pragma unroll
            for (int s = 0; s < 8; ++s) {
                const bf16* vrow =
                    vb + (size_t)(wv * 128 + s * 16 + l16) * TK
                       + kt + kc * 32 + hi * 8;
                BV av;  // A = V^T rows: halves hi*8..+7 and 16+hi*8..+7
                av.u[0] = *(const uint4*)(vrow);
                av.u[1] = *(const uint4*)(vrow + 16);
                oacc[s] = __builtin_amdgcn_wmma_f32_16x16x32_bf16(
                    false, av.v, false, bp.v, (short)0, oacc[s], false, false);
            }
        }
        __syncthreads();   // protect pbuf/red* for next iteration
    }

    // ----- epilogue: normalize, vectorized b128 stores -------------------
    float invl = 1.0f / l_run;
#pragma unroll
    for (int s = 0; s < 8; ++s) {
        float4 o0, o1;
        o0.x = oacc[s][0] * invl; o0.y = oacc[s][1] * invl;
        o0.z = oacc[s][2] * invl; o0.w = oacc[s][3] * invl;
        o1.x = oacc[s][4] * invl; o1.y = oacc[s][5] * invl;
        o1.z = oacc[s][6] * invl; o1.w = oacc[s][7] * invl;
        float* op = ob + (size_t)l16 * DDIM + wv * 128 + s * 16 + 8 * hi;
        *(float4*)(op)     = o0;
        *(float4*)(op + 4) = o1;
    }
}

extern "C" void kernel_launch(void* const* d_in, const int* in_sizes, int n_in,
                              void* d_out, int out_size, void* d_ws, size_t ws_size,
                              hipStream_t stream) {
    (void)in_sizes; (void)n_in; (void)out_size; (void)ws_size;
    const float* q = (const float*)d_in[0];
    const float* k = (const float*)d_in[1];
    const float* v = (const float*)d_in[2];
    const unsigned char* mask = (const unsigned char*)d_in[3]; // numpy bool_
    float* out = (float*)d_out;

    // Workspace layout: kT bf16 [B][Tk][D] then vT bf16 [B][D][Tk] (128 MB).
    bf16* kT = (bf16*)d_ws;
    bf16* vT = kT + (size_t)NB * TK * DDIM;

    // Pass 1: k [B][D][Tk] -> kT [B][Tk][D]; v [B][Tk][D] -> vT [B][D][Tk]
    transpose_cvt_kernel<<<dim3(TK / 64, DDIM / 64, NB), 256, 0, stream>>>(
        k, kT, DDIM, TK);
    transpose_cvt_kernel<<<dim3(DDIM / 64, TK / 64, NB), 256, 0, stream>>>(
        v, vT, TK, DDIM);

    // Pass 2: attention
    fa_fwd_kernel<<<dim3(TQ / 16, NB), 256, 0, stream>>>(q, kT, vT, mask, out);
}
